// SelfAttention_11373073400269
// MI455X (gfx1250) — matrix operand order repaired and checked by reading, not compiled, not run on hardware
//
#include <hip/hip_runtime.h>

#define BATCHN 2
#define SEQL   2048
#define EMB    2048
#define EMB3   6144
#define NHEAD  16
#define HDIM   128
#define NTOK   (BATCHN * SEQL) /* 4096 */

typedef __bf16 bf16_t;
typedef bf16_t bf16x8 __attribute__((ext_vector_type(8)));
typedef bf16_t v16bf  __attribute__((ext_vector_type(16)));
typedef float  v8f    __attribute__((ext_vector_type(8)));
typedef unsigned int u32x4 __attribute__((ext_vector_type(4)));
typedef int i32x4 __attribute__((ext_vector_type(4)));
typedef int i32x8 __attribute__((ext_vector_type(8)));

__device__ __forceinline__ unsigned short f32_to_bf16_bits(float f) {
  union { float f; unsigned int u; } c; c.f = f;
  unsigned int u = c.u;
  u += 0x7FFFu + ((u >> 16) & 1u);   // round-to-nearest-even
  return (unsigned short)(u >> 16);
}

__device__ __forceinline__ v8f wmma_bf16(v16bf a, v16bf b, v8f c) {
  return __builtin_amdgcn_wmma_f32_16x16x32_bf16(false, a, false, b,
                                                 (short)0, c, false, false);
}

// A fragment: 16x32 bf16, row-major, ld in elements (ISA 7.12.2 layout).
__device__ __forceinline__ v16bf load_a_frag(const bf16_t* base, int ld, int lane) {
  const int r = lane & 15, hi = lane >> 4;
  const bf16_t* p = base + (size_t)r * ld + hi * 8;
  bf16x8 c0 = *(const bf16x8*)(p);
  bf16x8 c1 = *(const bf16x8*)(p + 16);
  v16bf f;
#pragma unroll
  for (int i = 0; i < 8; ++i) { f[i] = c0[i]; f[8 + i] = c1[i]; }
  return f;
}

// B fragment: 32x16 (KxN) where B[k][n] = mem[n*ld + k].
__device__ __forceinline__ v16bf load_b_frag(const bf16_t* base, int ld, int lane) {
  const int n = lane & 15, hi = lane >> 4;
  const bf16_t* p = base + (size_t)n * ld + hi * 8;
  bf16x8 c0 = *(const bf16x8*)(p);
  bf16x8 c1 = *(const bf16x8*)(p + 16);
  v16bf f;
#pragma unroll
  for (int i = 0; i < 8; ++i) { f[i] = c0[i]; f[8 + i] = c1[i]; }
  return f;
}

// ---- TDM: 2D tile global->LDS (D# per CDNA5 ISA) ----
__device__ __forceinline__ void tdm_load_2d(unsigned lds_addr, const void* gptr,
                                            unsigned tensor_d0, unsigned tensor_d1,
                                            unsigned tile_d0, unsigned tile_d1,
                                            unsigned long long stride0_elems) {
  unsigned long long ga = (unsigned long long)(size_t)gptr;
  u32x4 g0;
  g0[0] = 1u;                                          // count=1, user mode
  g0[1] = lds_addr;                                    // lds_addr [63:32]
  g0[2] = (unsigned)ga;                                // global_addr [95:64]
  g0[3] = (unsigned)((ga >> 32) & 0x01FFFFFFull)       // global_addr [120:96]
          | (2u << 30);                                // type=2 ("image")
  i32x8 g1;
  g1[0] = (int)(1u << 16);                             // data_size=1 (2B), mask=0
  g1[1] = (int)((tensor_d0 & 0xFFFFu) << 16);          // tensor_dim0[15:0]
  g1[2] = (int)(((tensor_d0 >> 16) & 0xFFFFu) |
                ((tensor_d1 & 0xFFFFu) << 16));        // td0[31:16] | td1[15:0]
  g1[3] = (int)(((tensor_d1 >> 16) & 0xFFFFu) |
                ((tile_d0 & 0xFFFFu) << 16));          // td1[31:16] | tile_dim0
  g1[4] = (int)(tile_d1 & 0xFFFFu);                    // tile_dim1, tile_dim2=0
  g1[5] = (int)(unsigned)(stride0_elems & 0xFFFFFFFFull);  // dim0_stride[31:0]
  g1[6] = (int)(unsigned)((stride0_elems >> 32) & 0xFFFFull);
  g1[7] = 0;
  i32x4 z4 = {0, 0, 0, 0};
#if defined(__clang_major__) && (__clang_major__ >= 23)
  i32x8 z8 = {0, 0, 0, 0, 0, 0, 0, 0};
  __builtin_amdgcn_tensor_load_to_lds(g0, g1, z4, z4, z8, 0);
#else
  __builtin_amdgcn_tensor_load_to_lds(g0, g1, z4, z4, 0);
#endif
}

// ---------------------------------------------------------------- f32 -> bf16
__global__ void cvt_bf16_kernel(const float* __restrict__ in,
                                unsigned short* __restrict__ out, int n) {
  int i = blockIdx.x * blockDim.x + threadIdx.x;
  int stride = gridDim.x * blockDim.x;
  for (; i < n; i += stride) out[i] = f32_to_bf16_bits(in[i]);
}

// ------------------------------------------------- QKV GEMM + bias + scatter
__global__ __launch_bounds__(256) void qkv_gemm_kernel(
    const unsigned short* __restrict__ Xb, const unsigned short* __restrict__ Wb,
    const float* __restrict__ b_in, unsigned short* __restrict__ Qb,
    unsigned short* __restrict__ Kb, unsigned short* __restrict__ Vt) {
  const int lane = threadIdx.x & 31;
  const int wave = blockIdx.x * (blockDim.x >> 5) + (threadIdx.x >> 5);
  const int TN = EMB3 / 64;                       // 96
  const int TM = NTOK / 32;                       // 128
  if (wave >= TM * TN) return;
  const int m0 = (wave / TN) * 32, n0 = (wave % TN) * 64;

  const bf16_t* A = (const bf16_t*)Xb;
  const bf16_t* B = (const bf16_t*)Wb;

  v8f acc[2][4];
#pragma unroll
  for (int mi = 0; mi < 2; ++mi)
#pragma unroll
    for (int j = 0; j < 4; ++j)
#pragma unroll
      for (int i = 0; i < 8; ++i) acc[mi][j][i] = 0.0f;

  for (int kk = 0; kk < EMB; kk += 32) {
    if (kk + 64 < EMB) {
      __builtin_prefetch((const void*)(A + (size_t)m0 * EMB + kk + 64), 0, 1);
      __builtin_prefetch((const void*)(B + (size_t)n0 * EMB + kk + 64), 0, 1);
    }
    v16bf a0 = load_a_frag(A + (size_t)m0 * EMB + kk, EMB, lane);
    v16bf a1 = load_a_frag(A + (size_t)(m0 + 16) * EMB + kk, EMB, lane);
#pragma unroll
    for (int j = 0; j < 4; ++j) {
      v16bf bf = load_b_frag(B + (size_t)(n0 + j * 16) * EMB + kk, EMB, lane);
      acc[0][j] = wmma_bf16(a0, bf, acc[0][j]);
      acc[1][j] = wmma_bf16(a1, bf, acc[1][j]);
    }
  }

  const int hi = lane >> 4, nlo = lane & 15;
#pragma unroll
  for (int mi = 0; mi < 2; ++mi)
#pragma unroll
    for (int j = 0; j < 4; ++j)
#pragma unroll
      for (int i = 0; i < 8; ++i) {
        int row = m0 + mi * 16 + hi * 8 + i;
        int col = n0 + j * 16 + nlo;
        unsigned short bb = f32_to_bf16_bits(acc[mi][j][i] + b_in[col]);
        int part = col >> 11;
        int e = col & 2047;
        int h = e >> 7, d = e & 127;
        int b = row >> 11, l = row & 2047;
        int bh = b * NHEAD + h;
        if (part == 0)      Qb[((size_t)bh * SEQL + l) * HDIM + d] = bb;
        else if (part == 1) Kb[((size_t)bh * SEQL + l) * HDIM + d] = bb;
        else                Vt[((size_t)bh * HDIM + d) * SEQL + l] = bb;
      }
}

// ----------------------------------------------------- flash attention (fwd)
// Block = 8 waves = 8 consecutive 16-row q tiles of one (b,h). K/V tiles are
// TDM-loaded into double-buffered LDS by wave 0 and shared by all 8 waves.
__global__ __launch_bounds__(256) void attn_kernel(
    const unsigned short* __restrict__ Qb, const unsigned short* __restrict__ Kb,
    const unsigned short* __restrict__ Vt, unsigned short* __restrict__ Ob,
    const int* __restrict__ causal_flag) {
  __shared__ __align__(16) unsigned short Ktile[2][32 * HDIM];   // 2 x 8 KiB
  __shared__ __align__(16) unsigned short Vtile[2][HDIM * 32];   // 2 x 8 KiB
  __shared__ __align__(16) unsigned short Pl[8][16 * 32];        // 8 KiB
  const int lane = threadIdx.x & 31;
  const int wslot = threadIdx.x >> 5;
  const int blk = blockIdx.x;                     // 512 blocks
  const int bh = blk >> 4;                        // 16 blocks per (b,h)
  const int qt8 = blk & 15;                       // q-tile group in (b,h)
  const int q0 = (qt8 * 8 + wslot) * 16;
  const int b = bh >> 4, h = bh & 15;
  const int causal = causal_flag[0];

  const bf16_t* Qp = (const bf16_t*)Qb + (size_t)bh * SEQL * HDIM;
  const bf16_t* Kp = (const bf16_t*)Kb + (size_t)bh * SEQL * HDIM;
  const bf16_t* Vp = (const bf16_t*)Vt + (size_t)bh * HDIM * SEQL;

  v16bf qa[4];
#pragma unroll
  for (int j = 0; j < 4; ++j)
    qa[j] = load_a_frag(Qp + (size_t)q0 * HDIM + j * 32, HDIM, lane);

  v8f o[8];
  float mrow[8], lrow[8];
#pragma unroll
  for (int f = 0; f < 8; ++f) {
#pragma unroll
    for (int i = 0; i < 8; ++i) o[f][i] = 0.0f;
    mrow[f] = -3.0e38f; lrow[f] = 0.0f;
  }

  const float sc = 0.08838834764831845f;          // 1/sqrt(128)
  const int hi = lane >> 4, nlo = lane & 15;
  // Block-uniform step count (largest q tile in block); shorter waves run
  // fully-masked steps which are exact no-ops (p=0, rescale factor=1).
  const int nkt = causal ? (qt8 * 4 + 4) : (SEQL >> 5);
  bf16_t* Pw = (bf16_t*)&Pl[wslot][0];

  if (wslot == 0) {                               // prologue: stage tile 0
    tdm_load_2d((unsigned)(size_t)&Ktile[0][0], Kp, HDIM, SEQL, HDIM, 32, HDIM);
    tdm_load_2d((unsigned)(size_t)&Vtile[0][0], Vp, SEQL, HDIM, 32, HDIM, SEQL);
  }

  for (int kt = 0; kt < nkt; ++kt) {
    const int kb = kt * 32;
    const int buf = kt & 1;
    if (wslot == 0) __builtin_amdgcn_s_wait_tensorcnt(0);
    __syncthreads();                              // tile kt visible to block
    if (wslot == 0 && kt + 1 < nkt) {             // prefetch tile kt+1
      const int kb1 = kb + 32, b1 = buf ^ 1;
      tdm_load_2d((unsigned)(size_t)&Ktile[b1][0], Kp + (size_t)kb1 * HDIM,
                  HDIM, SEQL, HDIM, 32, HDIM);
      tdm_load_2d((unsigned)(size_t)&Vtile[b1][0], Vp + kb1,
                  SEQL, HDIM, 32, HDIM, SEQL);
    }
    const bf16_t* Kl = (const bf16_t*)&Ktile[buf][0];   // [32][128]
    const bf16_t* Vl = (const bf16_t*)&Vtile[buf][0];   // [128][32]

    v8f s0, s1;
#pragma unroll
    for (int i = 0; i < 8; ++i) { s0[i] = 0.0f; s1[i] = 0.0f; }
#pragma unroll
    for (int j = 0; j < 4; ++j) {                 // S = Q @ K^T
      v16bf k0 = load_b_frag(Kl + j * 32, HDIM, lane);
      v16bf k1 = load_b_frag(Kl + 16 * HDIM + j * 32, HDIM, lane);
      s0 = wmma_bf16(qa[j], k0, s0);
      s1 = wmma_bf16(qa[j], k1, s1);
    }

    float p0[8], p1[8], fct[8];
#pragma unroll
    for (int i = 0; i < 8; ++i) {
      int qg = q0 + hi * 8 + i;
      float a0 = (!causal || (kb + nlo)      <= qg) ? s0[i] * sc : -3.0e38f;
      float a1 = (!causal || (kb + 16 + nlo) <= qg) ? s1[i] * sc : -3.0e38f;
      float mx = fmaxf(a0, a1);
      mx = fmaxf(mx, __shfl_xor(mx, 1, 32));
      mx = fmaxf(mx, __shfl_xor(mx, 2, 32));
      mx = fmaxf(mx, __shfl_xor(mx, 4, 32));
      mx = fmaxf(mx, __shfl_xor(mx, 8, 32));
      float mn = fmaxf(mrow[i], mx);
      float f  = __expf(mrow[i] - mn);
      p0[i] = __expf(a0 - mn);
      p1[i] = __expf(a1 - mn);
      float rs = p0[i] + p1[i];
      rs += __shfl_xor(rs, 1, 32);
      rs += __shfl_xor(rs, 2, 32);
      rs += __shfl_xor(rs, 4, 32);
      rs += __shfl_xor(rs, 8, 32);
      lrow[i] = lrow[i] * f + rs;
      mrow[i] = mn;
      fct[i] = f;
    }
#pragma unroll
    for (int f = 0; f < 8; ++f)
#pragma unroll
      for (int i = 0; i < 8; ++i) o[f][i] *= fct[i];

    // Stage P (C-layout) -> LDS row-major -> reload as A fragment.
#pragma unroll
    for (int i = 0; i < 8; ++i) {
      ((unsigned short*)Pw)[(hi * 8 + i) * 32 + nlo]      = f32_to_bf16_bits(p0[i]);
      ((unsigned short*)Pw)[(hi * 8 + i) * 32 + 16 + nlo] = f32_to_bf16_bits(p1[i]);
    }
    v16bf pa = load_a_frag(Pw, 32, lane);

#pragma unroll
    for (int f = 0; f < 8; ++f) {                 // O += P @ V
      v16bf vb = load_b_frag(Vl + (size_t)(f * 16) * 32, 32, lane);
      o[f] = wmma_bf16(pa, vb, o[f]);
    }
    __syncthreads();                              // done reading buf
  }

  float inv[8];
#pragma unroll
  for (int i = 0; i < 8; ++i) inv[i] = 1.0f / lrow[i];
#pragma unroll
  for (int f = 0; f < 8; ++f)
#pragma unroll
    for (int i = 0; i < 8; ++i) {
      int row = q0 + hi * 8 + i;
      int t = b * SEQL + row;
      int e = h * HDIM + f * 16 + nlo;
      Ob[(size_t)t * EMB + e] = f32_to_bf16_bits(o[f][i] * inv[i]);
    }
}

// ------------------------------------------- output GEMM + bias (f32 result)
__global__ __launch_bounds__(256) void out_gemm_kernel(
    const unsigned short* __restrict__ Ob, const unsigned short* __restrict__ Wb,
    const float* __restrict__ b_out, float* __restrict__ out) {
  const int lane = threadIdx.x & 31;
  const int wave = blockIdx.x * (blockDim.x >> 5) + (threadIdx.x >> 5);
  const int TN = EMB / 64;                        // 32
  const int TM = NTOK / 32;                       // 128
  if (wave >= TM * TN) return;
  const int m0 = (wave / TN) * 32, n0 = (wave % TN) * 64;

  const bf16_t* A = (const bf16_t*)Ob;
  const bf16_t* B = (const bf16_t*)Wb;

  v8f acc[2][4];
#pragma unroll
  for (int mi = 0; mi < 2; ++mi)
#pragma unroll
    for (int j = 0; j < 4; ++j)
#pragma unroll
      for (int i = 0; i < 8; ++i) acc[mi][j][i] = 0.0f;

  for (int kk = 0; kk < EMB; kk += 32) {
    if (kk + 64 < EMB) {
      __builtin_prefetch((const void*)(A + (size_t)m0 * EMB + kk + 64), 0, 1);
      __builtin_prefetch((const void*)(B + (size_t)n0 * EMB + kk + 64), 0, 1);
    }
    v16bf a0 = load_a_frag(A + (size_t)m0 * EMB + kk, EMB, lane);
    v16bf a1 = load_a_frag(A + (size_t)(m0 + 16) * EMB + kk, EMB, lane);
#pragma unroll
    for (int j = 0; j < 4; ++j) {
      v16bf bf = load_b_frag(B + (size_t)(n0 + j * 16) * EMB + kk, EMB, lane);
      acc[0][j] = wmma_bf16(a0, bf, acc[0][j]);
      acc[1][j] = wmma_bf16(a1, bf, acc[1][j]);
    }
  }

  const int hi = lane >> 4, nlo = lane & 15;
#pragma unroll
  for (int mi = 0; mi < 2; ++mi)
#pragma unroll
    for (int j = 0; j < 4; ++j)
#pragma unroll
      for (int i = 0; i < 8; ++i) {
        int row = m0 + mi * 16 + hi * 8 + i;
        int col = n0 + j * 16 + nlo;
        out[(size_t)row * EMB + col] = acc[mi][j][i] + b_out[col];
      }
}

extern "C" void kernel_launch(void* const* d_in, const int* in_sizes, int n_in,
                              void* d_out, int out_size, void* d_ws, size_t ws_size,
                              hipStream_t stream) {
  const float* x      = (const float*)d_in[0];
  const float* w_in   = (const float*)d_in[1];
  const float* b_in   = (const float*)d_in[2];
  const float* w_out  = (const float*)d_in[3];
  const float* b_out  = (const float*)d_in[4];
  const int*   causal = (const int*)d_in[5];

  char* ws = (char*)d_ws;
  unsigned short* Xb    = (unsigned short*)(ws + 0);            // 16 MiB
  unsigned short* Winb  = (unsigned short*)(ws + 16777216);     // 24 MiB
  unsigned short* Woutb = (unsigned short*)(ws + 41943040);     //  8 MiB
  unsigned short* Qb    = (unsigned short*)(ws + 50331648);     // 16 MiB
  unsigned short* Kb    = (unsigned short*)(ws + 67108864);     // 16 MiB
  unsigned short* Vt    = (unsigned short*)(ws + 83886080);     // 16 MiB
  unsigned short* Ob    = (unsigned short*)(ws + 100663296);    // 16 MiB

  auto cvt = [&](const float* in, unsigned short* out, int n) {
    int blocks = (n + 255) / 256; if (blocks > 4096) blocks = 4096;
    cvt_bf16_kernel<<<blocks, 256, 0, stream>>>(in, out, n);
  };
  cvt(x,     Xb,    NTOK * EMB);
  cvt(w_in,  Winb,  EMB3 * EMB);
  cvt(w_out, Woutb, EMB  * EMB);

  {
    int waves = (NTOK / 32) * (EMB3 / 64);        // 12288
    qkv_gemm_kernel<<<waves / 8, 256, 0, stream>>>(Xb, Winb, b_in, Qb, Kb, Vt);
  }
  {
    attn_kernel<<<BATCHN * NHEAD * 16, 256, 0, stream>>>(Qb, Kb, Vt, Ob, causal);
  }
  {
    int waves = (NTOK / 32) * (EMB / 64);         // 4096
    out_gemm_kernel<<<waves / 8, 256, 0, stream>>>(Ob, Woutb, b_out, (float*)d_out);
  }
}
